// GCN_13572096655678
// MI455X (gfx1250) — matrix operand order repaired and checked by reading, not compiled, not run on hardware
//
#include <hip/hip_runtime.h>
#include <hip/hip_bf16.h>

// ---------------------------------------------------------------------------
// NNConv x2 for MI455X (gfx1250, wave32, WMMA).
//
// msg[e,o] = sum_{k,i} (h[e,k] * x_src[e,i]) * B[(k,i),o]
//   where B[(k,i),o] = w2[k, i*64+o]  for k<64,   B[(64,i),o] = b2[i*64+o]
// -> one GEMM [E, 65*in] x [65*in, 64], A built on the fly as outer products.
// ---------------------------------------------------------------------------

#define NNODES 20000
#define NEDGES 50000
#define INC0   32
#define HDIM   64

typedef __attribute__((ext_vector_type(16))) __bf16 v16bf;
typedef __attribute__((ext_vector_type(8)))  float  v8f;

static __device__ __forceinline__ __bf16 f2bf(float f) { return (__bf16)f; }

// ---------------------------------------------------------------------------
// Build WMMA-fragment-ordered bf16 B buffer from w2 [64, INC*64] and b2 [INC*64].
// Layout: Bsh[kb][ns][lane][j] ; element = B[q = kb*32 + 8*(lane/16) + j + 8*(j>=8), o = ns*16 + lane%16]
// (16-bit B 32x16 fragment: lane%16 <-> N, lane half and j <-> K in AMD table order.)
// ---------------------------------------------------------------------------
template<int INC>
__global__ void build_B_kernel(const float* __restrict__ w2,
                               const float* __restrict__ b2,
                               unsigned short* __restrict__ Bsh)
{
    constexpr int KB = (65 * INC) / 32;
    const int total = KB * 4 * 32 * 16;
    int idx = blockIdx.x * blockDim.x + threadIdx.x;
    if (idx >= total) return;
    int j  = idx & 15;
    int L  = (idx >> 4) & 31;
    int ns = (idx >> 9) & 3;
    int kb = idx >> 11;
    int q = kb * 32 + 8 * (L >> 4) + j + ((j >= 8) ? 8 : 0);
    int t = q / INC;
    int i = q - t * INC;
    int o = ns * 16 + (L & 15);
    float v = (t < 64) ? w2[t * (INC * 64) + i * 64 + o] : b2[i * 64 + o];
    union { __bf16 b; unsigned short s; } cv;
    cv.b = f2bf(v);
    Bsh[idx] = cv.s;
}

// ---------------------------------------------------------------------------
// Edge MLP stage 1: h[e,o] = relu(ea[e,0]*w1[0,o] + ea[e,1]*w1[1,o] + b1[o])
// ---------------------------------------------------------------------------
__global__ void edge_mlp1_kernel(const float* __restrict__ ea,
                                 const float* __restrict__ w1,
                                 const float* __restrict__ b1,
                                 float* __restrict__ hbuf, int E)
{
    int idx = blockIdx.x * blockDim.x + threadIdx.x;
    if (idx >= E * 64) return;
    int e = idx >> 6;
    int o = idx & 63;
    float a0 = ea[e * 2 + 0];
    float a1 = ea[e * 2 + 1];
    float v = fmaf(a0, w1[o], fmaf(a1, w1[64 + o], b1[o]));
    hbuf[idx] = fmaxf(v, 0.0f);
}

// ---------------------------------------------------------------------------
// agg[n,o] = bias[o] + sum_i x[n,i] * root[i,o]   (root/self term, pre-scatter)
// ---------------------------------------------------------------------------
template<int INC>
__global__ void node_root_bias_kernel(const float* __restrict__ xin,
                                      const float* __restrict__ root,
                                      const float* __restrict__ bias,
                                      float* __restrict__ agg, int N)
{
    int idx = blockIdx.x * blockDim.x + threadIdx.x;
    if (idx >= N * 64) return;
    int n = idx >> 6;
    int o = idx & 63;
    float acc = bias[o];
    const float* xr = xin + n * INC;
#pragma unroll 8
    for (int i = 0; i < INC; ++i)
        acc = fmaf(xr[i], root[i * 64 + o], acc);
    agg[idx] = acc;
}

// ---------------------------------------------------------------------------
// Fused outer-product WMMA GEMM + scatter-add.
// Tile: 32 edges (M) x 64 outputs (N), 4 waves (one 16-wide N strip each),
// K = 65*INC, A built on the fly: A[m, t*INC+i] = coef(m,t) * x_src[m,i],
// coef = h[m,t] (t<64) or 1.0 (t==64, bias rows of B).
// B fragments are software-pipelined (next-iter prefetch into registers).
// ---------------------------------------------------------------------------
template<int INC>
__global__ __launch_bounds__(128)
void edge_gemm_wmma_kernel(const float* __restrict__ xin,          // [N, INC]
                           const float* __restrict__ hbuf,         // [E, 64]
                           const unsigned short* __restrict__ Bsh, // fragment order (+4KB pad)
                           const int* __restrict__ srcIdx,
                           const int* __restrict__ dstIdx,
                           float* __restrict__ agg,                // [N, 64]
                           int E)
{
    constexpr int NHALF = INC / 32;        // K sub-blocks per t
    __shared__ float h_s[32][65];          // padded vs. bank conflicts
    __shared__ float x_s[32][INC + 1];
    __shared__ int   dst_s[32];

    const int tid  = threadIdx.x;
    const int lane = tid & 31;
    const int wave = tid >> 5;             // N strip 0..3
    const int e0   = blockIdx.x * 32;

    if (tid < 32) {
        int e = e0 + tid;
        dst_s[tid] = (e < E) ? dstIdx[e] : -1;
    }
    for (int i = tid; i < 32 * 64; i += 128) {
        int m = i >> 6, o = i & 63;
        int e = e0 + m;
        h_s[m][o] = (e < E) ? hbuf[e * 64 + o] : 0.0f;
    }
    for (int i = tid; i < 32 * INC; i += 128) {
        int m = i / INC, c = i - m * INC;
        int e = e0 + m;
        float v = 0.0f;
        if (e < E) v = xin[srcIdx[e] * INC + c];
        x_s[m][c] = v;
    }
    __syncthreads();

    const int m0 = lane & 15;
    const int m1 = m0 + 16;
    const int kbase = (lane >= 16) ? 8 : 0;

    // Hoist x values this lane needs: same 16 K-slots every block (ISA A layout:
    // lane->M, j -> K = kbase + j + 8*(j>=8)).
    float xr0[NHALF][16];
    float xr1[NHALF][16];
#pragma unroll
    for (int hh = 0; hh < NHALF; ++hh) {
#pragma unroll
        for (int j = 0; j < 16; ++j) {
            int k = hh * 32 + kbase + j + ((j >= 8) ? 8 : 0);
            xr0[hh][j] = x_s[m0][k];
            xr1[hh][j] = x_s[m1][k];
        }
    }

    v8f c0 = {};
    v8f c1 = {};
    const unsigned short* bptr = Bsh + (wave * 32 + lane) * 16;
    v16bf bcur = *(const v16bf*)bptr;      // first fragment

    for (int t = 0; t < 65; ++t) {
        float coef0 = (t < 64) ? h_s[m0][t] : 1.0f;
        float coef1 = (t < 64) ? h_s[m1][t] : 1.0f;
#pragma unroll
        for (int hh = 0; hh < NHALF; ++hh) {
            bptr += 4 * 32 * 16;
            v16bf bnext = *(const v16bf*)bptr;   // prefetch next (pad covers last)
            v16bf a0, a1;
#pragma unroll
            for (int j = 0; j < 16; ++j) {
                a0[j] = f2bf(coef0 * xr0[hh][j]);
                a1[j] = f2bf(coef1 * xr1[hh][j]);
            }
            c0 = __builtin_amdgcn_wmma_f32_16x16x32_bf16(
                     false, a0, false, bcur, (short)0, c0, false, false);
            c1 = __builtin_amdgcn_wmma_f32_16x16x32_bf16(
                     false, a1, false, bcur, (short)0, c1, false, false);
            bcur = bnext;
        }
    }

    // Scatter: C layout lane -> N = lane&15, vgpr r -> M = r + 8*(lane>=16).
    const int ncol = wave * 16 + (lane & 15);
    const int mhi  = (lane >> 4) * 8;
    if (e0 + 32 <= E) {
        // Full tile: dst indices always valid -> unguarded atomics, no branches.
#pragma unroll
        for (int r = 0; r < 8; ++r) {
            int mA = r + mhi;
            atomicAdd(&agg[dst_s[mA]      * 64 + ncol], c0[r]);
            atomicAdd(&agg[dst_s[mA + 16] * 64 + ncol], c1[r]);
        }
    } else {
#pragma unroll
        for (int r = 0; r < 8; ++r) {
            int mA = r + mhi;
            int d0 = dst_s[mA];
            if (d0 >= 0) atomicAdd(&agg[d0 * 64 + ncol], c0[r]);
            int d1 = dst_s[mA + 16];
            if (d1 >= 0) atomicAdd(&agg[d1 * 64 + ncol], c1[r]);
        }
    }
}

__global__ void relu_kernel(const float* __restrict__ in, float* __restrict__ out, int n)
{
    int i = blockIdx.x * blockDim.x + threadIdx.x;
    if (i < n) out[i] = fmaxf(in[i], 0.0f);
}

// ---------------------------------------------------------------------------
extern "C" void kernel_launch(void* const* d_in, const int* in_sizes, int n_in,
                              void* d_out, int out_size, void* d_ws, size_t ws_size,
                              hipStream_t stream)
{
    const float* x      = (const float*)d_in[0];
    const int*   eidx   = (const int*)  d_in[1];
    const float* eattr  = (const float*)d_in[2];
    const float* w1_0   = (const float*)d_in[3];
    const float* b1_0   = (const float*)d_in[4];
    const float* w2_0   = (const float*)d_in[5];
    const float* b2_0   = (const float*)d_in[6];
    const float* root_0 = (const float*)d_in[7];
    const float* bias_0 = (const float*)d_in[8];
    const float* w1_1   = (const float*)d_in[9];
    const float* b1_1   = (const float*)d_in[10];
    const float* w2_1   = (const float*)d_in[11];
    const float* b2_1   = (const float*)d_in[12];
    const float* root_1 = (const float*)d_in[13];
    const float* bias_1 = (const float*)d_in[14];

    const int N = NNODES, E = NEDGES;
    const int* srcIdx = eidx;         // edge_index[0]
    const int* dstIdx = eidx + E;     // edge_index[1]

    // Workspace carve-up (B buffers padded by one fragment row = 4KB for the
    // software-pipeline's one-past-end prefetch).
    char* ws = (char*)d_ws;
    float* hbuf = (float*)ws;                   ws += (size_t)E * 64 * 4;            // 12.8 MB
    float* agg  = (float*)ws;                   ws += (size_t)N * 64 * 4;            // 5.12 MB
    float* x1   = (float*)ws;                   ws += (size_t)N * 64 * 4;            // 5.12 MB
    unsigned short* Bsh0 = (unsigned short*)ws; ws += ((size_t)(65 * 32) * 64 + 2048) * 2;
    unsigned short* Bsh1 = (unsigned short*)ws; ws += ((size_t)(65 * 64) * 64 + 2048) * 2;
    (void)ws_size;

    const int TPB = 256;
    // --- per-call B reshuffles (tiny) ---
    {
        int tot0 = (65 * INC0 / 32) * 2048;
        build_B_kernel<INC0><<<(tot0 + TPB - 1) / TPB, TPB, 0, stream>>>(w2_0, b2_0, Bsh0);
        int tot1 = (65 * HDIM / 32) * 2048;
        build_B_kernel<HDIM><<<(tot1 + TPB - 1) / TPB, TPB, 0, stream>>>(w2_1, b2_1, Bsh1);
    }

    const int tiles = (E + 31) / 32;

    // ---------------- Layer 0: in=32 -> 64 ----------------
    edge_mlp1_kernel<<<(E * 64 + TPB - 1) / TPB, TPB, 0, stream>>>(eattr, w1_0, b1_0, hbuf, E);
    node_root_bias_kernel<INC0><<<(N * 64 + TPB - 1) / TPB, TPB, 0, stream>>>(x, root_0, bias_0, agg, N);
    edge_gemm_wmma_kernel<INC0><<<tiles, 128, 0, stream>>>(x, hbuf, Bsh0, srcIdx, dstIdx, agg, E);
    relu_kernel<<<(N * 64 + TPB - 1) / TPB, TPB, 0, stream>>>(agg, x1, N * 64);

    // ---------------- Layer 1: in=64 -> 64 ----------------
    edge_mlp1_kernel<<<(E * 64 + TPB - 1) / TPB, TPB, 0, stream>>>(eattr, w1_1, b1_1, hbuf, E);
    node_root_bias_kernel<HDIM><<<(N * 64 + TPB - 1) / TPB, TPB, 0, stream>>>(x1, root_1, bias_1, agg, N);
    edge_gemm_wmma_kernel<HDIM><<<tiles, 128, 0, stream>>>(x1, hbuf, Bsh1, srcIdx, dstIdx, agg, E);
    relu_kernel<<<(N * 64 + TPB - 1) / TPB, TPB, 0, stream>>>(agg, (float*)d_out, N * 64);
}
